// Attention_78185584657148
// MI455X (gfx1250) — matrix operand order repaired and checked by reading, not compiled, not run on hardware
//
#include <hip/hip_runtime.h>
#include <cstdint>

// ---------------------------------------------------------------------------
// BEiT attention: B=16, N=577, C=768, H=12, HD=64.  bf16 WMMA pipeline.
// ---------------------------------------------------------------------------
#define BATCH 16
#define SEQ   577
#define CH    768
#define NH    12
#define HD    64
#define NPAD  608              // 19 * 32
#define ROWS  (BATCH * SEQ)    // 9232 = 577 * 16  (divisible by 16)

typedef __attribute__((ext_vector_type(16))) __bf16 v16bf;
typedef __attribute__((ext_vector_type(8)))  float  v8f;

struct BfT {
    union { v16bf v; uint4 u[2]; };
};

__device__ __forceinline__ unsigned short f2bf(float f) {
    unsigned u = __builtin_bit_cast(unsigned, f);
    unsigned r = u + 0x7FFFu + ((u >> 16) & 1u);       // round-to-nearest-even
    if ((u & 0x7F800000u) == 0x7F800000u) r = u;       // inf/nan passthrough
    return (unsigned short)(r >> 16);
}

__device__ __forceinline__ v8f wmma_bf16(const BfT& a, const BfT& b, v8f c) {
    return __builtin_amdgcn_wmma_f32_16x16x32_bf16(
        /*neg_a=*/false, a.v, /*neg_b=*/false, b.v,
        /*c_mod=*/(short)0, c, /*reuse_a=*/false, /*reuse_b=*/false);
}

// reductions across the 16 lanes of each wave32 half (xor masks < 16 stay in-half)
__device__ __forceinline__ float rowmax16(float x) {
#pragma unroll
    for (int m = 8; m >= 1; m >>= 1) x = fmaxf(x, __shfl_xor(x, m, 32));
    return x;
}
__device__ __forceinline__ float rowsum16(float x) {
#pragma unroll
    for (int m = 8; m >= 1; m >>= 1) x += __shfl_xor(x, m, 32);
    return x;
}

// ---------------------------------------------------------------------------
// f32 -> bf16 convert
// ---------------------------------------------------------------------------
__global__ void cvt_kernel(const float* __restrict__ in,
                           unsigned short* __restrict__ out, int n) {
    int i = blockIdx.x * blockDim.x + threadIdx.x;
    if (i < n) out[i] = f2bf(in[i]);
}

// ---------------------------------------------------------------------------
// QKV GEMM: [9232,768] x [2304,768]^T, bias fused, scatter into head layouts.
// One wave per 16x16 D tile. grid = (144, 577)
// ---------------------------------------------------------------------------
__global__ __launch_bounds__(32)
void qkv_kernel(const unsigned short* __restrict__ xb,   // [9232,768] bf16
                const unsigned short* __restrict__ wb,   // [2304,768] bf16
                const float* __restrict__ q_bias,        // [768]
                const float* __restrict__ v_bias,        // [768]
                unsigned short* __restrict__ qo,         // [B*H, NPAD, 64]
                unsigned short* __restrict__ ko,         // [B*H, NPAD, 64]
                unsigned short* __restrict__ vto)        // [B*H, 64, NPAD]
{
    const int lane = threadIdx.x & 31;
    const int col  = lane & 15, half = lane >> 4;
    const int nt   = blockIdx.x;                 // 0..143
    const int mt   = blockIdx.y;                 // 0..576
    const int cout = nt * 16 + col;              // output column 0..2303

    const unsigned short* arow = xb + (size_t)(mt * 16 + col) * CH;
    const unsigned short* brow = wb + (size_t)cout * CH;

    v8f acc = {};
#pragma unroll 4
    for (int k0 = 0; k0 < CH; k0 += 32) {
        BfT a, b;
        a.u[0] = *(const uint4*)(arow + k0 + half * 16);
        a.u[1] = *(const uint4*)(arow + k0 + half * 16 + 8);
        b.u[0] = *(const uint4*)(brow + k0 + half * 16);
        b.u[1] = *(const uint4*)(brow + k0 + half * 16 + 8);
        acc = wmma_bf16(a, b, acc);
    }

    const int j  = cout / CH;        // 0=q 1=k 2=v
    const int cc = cout % CH;
    const int h  = cc >> 6, d = cc & 63;
    const float bias = (j == 0) ? q_bias[cc] : (j == 2 ? v_bias[cc] : 0.0f);
    const float mul  = (j == 0) ? 0.125f : 1.0f;   // HD^-0.5 folded into q

#pragma unroll
    for (int r = 0; r < 8; ++r) {
        const int m = mt * 16 + r + 8 * half;       // 0..9231
        const int b = m / SEQ, nn = m % SEQ;
        const unsigned short val = f2bf((acc[r] + bias) * mul);
        const size_t bh = (size_t)(b * NH + h);
        if (j == 0)      qo[(bh * NPAD + nn) * HD + d]  = val;
        else if (j == 1) ko[(bh * NPAD + nn) * HD + d]  = val;
        else             vto[(bh * HD + d) * NPAD + nn] = val;  // V transposed
    }
}

// ---------------------------------------------------------------------------
// Fused flash attention. One wave owns a 16-row q block for one (b,h);
// streams 32 keys/iter: 2x2 WMMAs for S, online softmax, P via LDS re-layout,
// 4 WMMAs for O.  grid = (37, 12, 16)
// ---------------------------------------------------------------------------
__global__ __launch_bounds__(32)
void attn_kernel(const unsigned short* __restrict__ qmat,  // [B*H, NPAD, 64]
                 const unsigned short* __restrict__ kmat,  // [B*H, NPAD, 64]
                 const unsigned short* __restrict__ vtm,   // [B*H, 64, NPAD]
                 const float* __restrict__ rel_bias,       // [H, 577, 577]
                 unsigned short* __restrict__ yo)          // [9232, 768] bf16
{
    __shared__ __align__(16) unsigned short lds_p[16 * 32];   // P tile 16x32

    const int lane = threadIdx.x & 31;
    const int col  = lane & 15, half = lane >> 4;
    const int qb = blockIdx.x;          // 0..36
    const int h  = blockIdx.y;
    const int b  = blockIdx.z;
    const size_t bh = (size_t)(b * NH + h);

    const unsigned short* qbase = qmat + bh * (size_t)NPAD * HD;
    const unsigned short* kbase = kmat + bh * (size_t)NPAD * HD;
    const unsigned short* vbase = vtm  + bh * (size_t)HD * NPAD;

    // q A-tiles (16x64 = two 16x32 tiles), resident for the whole loop
    BfT aLo, aHi;
    {
        const unsigned short* qr = qbase + (size_t)(qb * 16 + col) * HD;
        aLo.u[0] = *(const uint4*)(qr + half * 16);
        aLo.u[1] = *(const uint4*)(qr + half * 16 + 8);
        aHi.u[0] = *(const uint4*)(qr + 32 + half * 16);
        aHi.u[1] = *(const uint4*)(qr + 32 + half * 16 + 8);
    }

    v8f o[4]; o[0] = {}; o[1] = {}; o[2] = {}; o[3] = {};
    float mrow[8], lrow[8];
#pragma unroll
    for (int r = 0; r < 8; ++r) { mrow[r] = -INFINITY; lrow[r] = 0.0f; }

    for (int kb = 0; kb < NPAD / 32; ++kb) {
        const int key0 = kb * 32;

        // ---- S = (q*scale) @ k^T : two 16x16 tiles -------------------------
        v8f s[2];
#pragma unroll
        for (int t = 0; t < 2; ++t) {
            const unsigned short* kr =
                kbase + (size_t)(key0 + t * 16 + col) * HD;
            BfT bLo, bHi;
            bLo.u[0] = *(const uint4*)(kr + half * 16);
            bLo.u[1] = *(const uint4*)(kr + half * 16 + 8);
            bHi.u[0] = *(const uint4*)(kr + 32 + half * 16);
            bHi.u[1] = *(const uint4*)(kr + 32 + half * 16 + 8);
            v8f c = {};
            c = wmma_bf16(aLo, bLo, c);
            c = wmma_bf16(aHi, bHi, c);
            s[t] = c;
        }

        // ---- + rel_pos_bias, mask padded keys ------------------------------
#pragma unroll
        for (int t = 0; t < 2; ++t) {
            const int key = key0 + t * 16 + col;
#pragma unroll
            for (int r = 0; r < 8; ++r) {
                const int qrow = qb * 16 + r + 8 * half;
                float sv = s[t][r];
                if (key < SEQ) {
                    if (qrow < SEQ)
                        sv += rel_bias[((size_t)h * SEQ + qrow) * SEQ + key];
                } else {
                    sv = -INFINITY;
                }
                s[t][r] = sv;
            }
        }

        // ---- online softmax; write P (bf16) into LDS in C-layout -----------
#pragma unroll
        for (int r = 0; r < 8; ++r) {
            const float mnew = rowmax16(fmaxf(s[0][r], s[1][r]));
            const float mx   = fmaxf(mrow[r], mnew);
            const float f    = __expf(mrow[r] - mx);
            const float p0   = __expf(s[0][r] - mx);
            const float p1   = __expf(s[1][r] - mx);
            lrow[r] = lrow[r] * f + rowsum16(p0 + p1);
            mrow[r] = mx;
            o[0][r] *= f; o[1][r] *= f; o[2][r] *= f; o[3][r] *= f;
            const int row = r + 8 * half;
            lds_p[row * 32 + col]      = f2bf(p0);
            lds_p[row * 32 + 16 + col] = f2bf(p1);
        }

        // ---- re-read P in A-layout (16x32) ---------------------------------
        BfT pA;
        {
            const unsigned short* pr = lds_p + col * 32;
            pA.u[0] = *(const uint4*)(pr + half * 16);
            pA.u[1] = *(const uint4*)(pr + half * 16 + 8);
        }

        // ---- O += P @ V : 4 WMMAs (V^T rows are contiguous in keys) --------
#pragma unroll
        for (int nt = 0; nt < 4; ++nt) {
            const unsigned short* vr =
                vbase + (size_t)(nt * 16 + col) * NPAD + key0 + half * 16;
            BfT bV;
            bV.u[0] = *(const uint4*)(vr);
            bV.u[1] = *(const uint4*)(vr + 8);
            o[nt] = wmma_bf16(pA, bV, o[nt]);
        }
    }

    // ---- normalize and store y[b, qrow, h*64 + d] as bf16 ------------------
#pragma unroll
    for (int r = 0; r < 8; ++r) {
        const int qrow = qb * 16 + r + 8 * half;
        if (qrow < SEQ) {
            const float inv = 1.0f / lrow[r];
            const size_t base = ((size_t)b * SEQ + qrow) * CH + h * HD;
            yo[base + col]      = f2bf(o[0][r] * inv);
            yo[base + 16 + col] = f2bf(o[1][r] * inv);
            yo[base + 32 + col] = f2bf(o[2][r] * inv);
            yo[base + 48 + col] = f2bf(o[3][r] * inv);
        }
    }
}

// ---------------------------------------------------------------------------
// Projection GEMM: out[9232,768] f32 = y_bf16 @ proj_w^T + proj_b
// grid = (48, 577)
// ---------------------------------------------------------------------------
__global__ __launch_bounds__(32)
void proj_kernel(const unsigned short* __restrict__ yb,   // [9232,768] bf16
                 const unsigned short* __restrict__ wb,   // [768,768] bf16
                 const float* __restrict__ pbias,         // [768]
                 float* __restrict__ out)                 // [9232,768] f32
{
    const int lane = threadIdx.x & 31;
    const int col  = lane & 15, half = lane >> 4;
    const int nt   = blockIdx.x;                 // 0..47
    const int mt   = blockIdx.y;                 // 0..576

    const unsigned short* arow = yb + (size_t)(mt * 16 + col) * CH;
    const unsigned short* brow = wb + (size_t)(nt * 16 + col) * CH;

    v8f acc = {};
#pragma unroll 4
    for (int k0 = 0; k0 < CH; k0 += 32) {
        BfT a, b;
        a.u[0] = *(const uint4*)(arow + k0 + half * 16);
        a.u[1] = *(const uint4*)(arow + k0 + half * 16 + 8);
        b.u[0] = *(const uint4*)(brow + k0 + half * 16);
        b.u[1] = *(const uint4*)(brow + k0 + half * 16 + 8);
        acc = wmma_bf16(a, b, acc);
    }

    const int c = nt * 16 + col;
    const float pb = pbias[c];
#pragma unroll
    for (int r = 0; r < 8; ++r) {
        const int m = mt * 16 + r + 8 * half;
        out[(size_t)m * CH + c] = acc[r] + pb;
    }
}

// ---------------------------------------------------------------------------
extern "C" void kernel_launch(void* const* d_in, const int* in_sizes, int n_in,
                              void* d_out, int out_size, void* d_ws, size_t ws_size,
                              hipStream_t stream) {
    const float* x        = (const float*)d_in[0];   // [16,577,768]
    const float* rel_bias = (const float*)d_in[1];   // [1,12,577,577]
    const float* qkv_w    = (const float*)d_in[2];   // [2304,768]
    const float* q_bias   = (const float*)d_in[3];   // [768]
    const float* v_bias   = (const float*)d_in[4];   // [768]
    const float* proj_w   = (const float*)d_in[5];   // [768,768]
    const float* proj_b   = (const float*)d_in[6];   // [768]
    float* out            = (float*)d_out;

    // workspace layout (256B aligned chunks)
    char* ws = (char*)d_ws;
    size_t off = 0;
    auto alloc = [&](size_t bytes) {
        size_t o = off;
        off += (bytes + 255) & ~(size_t)255;
        return o;
    };
    const size_t o_xb  = alloc((size_t)ROWS * CH * 2);         // x bf16
    const size_t o_wq  = alloc((size_t)3 * CH * CH * 2);       // qkv_w bf16
    const size_t o_wp  = alloc((size_t)CH * CH * 2);           // proj_w bf16
    const size_t o_q   = alloc((size_t)BATCH * NH * NPAD * HD * 2);
    const size_t o_k   = alloc((size_t)BATCH * NH * NPAD * HD * 2);
    const size_t o_vt  = alloc((size_t)BATCH * NH * HD * NPAD * 2);
    const size_t o_y   = alloc((size_t)ROWS * CH * 2);
    (void)ws_size;

    unsigned short* xb  = (unsigned short*)(ws + o_xb);
    unsigned short* wqb = (unsigned short*)(ws + o_wq);
    unsigned short* wpb = (unsigned short*)(ws + o_wp);
    unsigned short* qb  = (unsigned short*)(ws + o_q);
    unsigned short* kb  = (unsigned short*)(ws + o_k);
    unsigned short* vtb = (unsigned short*)(ws + o_vt);
    unsigned short* yb  = (unsigned short*)(ws + o_y);

    // zero q/k/v^T (contiguous) so padded rows are clean
    hipMemsetAsync(ws + o_q, 0, (size_t)3 * BATCH * NH * NPAD * HD * 2, stream);

    // f32 -> bf16 converts
    {
        int n = ROWS * CH;
        cvt_kernel<<<(n + 255) / 256, 256, 0, stream>>>(x, xb, n);
        n = 3 * CH * CH;
        cvt_kernel<<<(n + 255) / 256, 256, 0, stream>>>(qkv_w, wqb, n);
        n = CH * CH;
        cvt_kernel<<<(n + 255) / 256, 256, 0, stream>>>(proj_w, wpb, n);
    }

    // QKV GEMM + scatter
    qkv_kernel<<<dim3(3 * CH / 16, ROWS / 16), 32, 0, stream>>>(
        xb, wqb, q_bias, v_bias, qb, kb, vtb);

    // fused flash attention
    attn_kernel<<<dim3((SEQ + 15) / 16, NH, BATCH), 32, 0, stream>>>(
        qb, kb, vtb, rel_bias, yb);

    // projection
    proj_kernel<<<dim3(CH / 16, ROWS / 16), 32, 0, stream>>>(
        yb, wpb, proj_b, out);
}
